// RNN_70935679861111
// MI455X (gfx1250) — compile-verified, matrix operand order checked by reading, not run on hardware
//
#include <hip/hip_runtime.h>
#include <hip/hip_bf16.h>

// LSTM head, persistent-workgroup recurrent kernel using CDNA5 WMMA bf16.
//
// Shapes (reference): x[B=256, T=2048, I=16], W_ih[256,16], W_hh[256,64],
// b_ih[256], b_hh[256], W_out[2,64], b_out[2], out[B, T, 2] fp32.
//
// Mapping:
//   grid = 16 blocks x 128 threads (4 waves of 32).
//   Block bb owns batch rows b0 = bb*16 .. b0+15 for the whole T loop.
//   Wave w (0..3) owns hidden columns j = w*16 .. w*16+15, i.e. the four
//   16x16 gate tiles at gate columns {0,64,128,192} + w*16.
//   Per step: gates = [h | x_pad] @ [W_hh^T ; W_ih^T_pad],  K = 32+32+32.
//   A (16x32 bf16) rebuilt from LDS each step; B (32x16 bf16) in registers.
//   Activations use v_exp_f32 + v_rcp_f32 (no IEEE-division slow path).

typedef __bf16  v16bf __attribute__((ext_vector_type(16)));
typedef __bf16  v8bf  __attribute__((ext_vector_type(8)));
typedef float   v8f   __attribute__((ext_vector_type(8)));

#define HXS 88          // LDS row stride in bf16 elems (mult of 8 -> 16B aligned chunks)
#define NH  64          // hidden size
#define BT  16          // batch tile per block

static __device__ __forceinline__ v16bf pack16(v8bf lo, v8bf hi) {
  v16bf r;
#pragma unroll
  for (int i = 0; i < 8; ++i) { r[i] = lo[i]; r[8 + i] = hi[i]; }
  return r;
}

static __device__ __forceinline__ float fast_sigmoid(float x) {
  // v_exp_f32 + v_rcp_f32, no div_scale/div_fmas sequence
  return __builtin_amdgcn_rcpf(1.0f + __expf(-x));
}
static __device__ __forceinline__ float fast_tanh(float x) {
  float e2 = __expf(2.0f * x);
  return (e2 - 1.0f) * __builtin_amdgcn_rcpf(e2 + 1.0f);
}

__global__ __launch_bounds__(128)
void lstm_head_wmma(const float* __restrict__ x,
                    const float* __restrict__ W_ih,
                    const float* __restrict__ W_hh,
                    const float* __restrict__ b_ih,
                    const float* __restrict__ b_hh,
                    const float* __restrict__ W_out,
                    const float* __restrict__ b_out,
                    float* __restrict__ out,
                    int T) {
  __shared__ __align__(16) __bf16 hx[BT * HXS];  // [row][col]: cols 0..63 = h, 64..79 = x
  __shared__ float outAcc[BT * 2];

  const int tid = threadIdx.x;
  const int w   = tid >> 5;          // wave 0..3
  const int lane = tid & 31;
  const int lg  = lane >> 4;         // lane group 0/1
  const int ln  = lane & 15;
  const int b0  = blockIdx.x * BT;

  // ---- preamble: zero LDS (h=0 initial state), zero out accumulators ----
  for (int e = tid; e < BT * HXS; e += 128) hx[e] = (__bf16)0.0f;
  if (tid < 32) outAcc[tid] = 0.0f;

  // ---- per-lane constants ----
  // gate bias for this wave's 4 tiles (column = G*64 + w*16 + ln)
  float bias4[4];
#pragma unroll
  for (int G = 0; G < 4; ++G) {
    int g = G * NH + w * 16 + ln;
    bias4[G] = b_ih[g] + b_hh[g];
  }
  // W_out slice for the output-partial phase: m = tid&15, o=(tid>>4)&1, q=tid>>5
  const int po = (tid >> 4) & 1;
  const int pq = tid >> 5;
  float wr[16];
#pragma unroll
  for (int i = 0; i < 16; ++i) wr[i] = W_out[po * NH + pq * 16 + i];
  float bo = (tid < 32) ? b_out[tid >> 4] : 0.0f;

  // ---- B-matrix fragments (constant over time): Bf[G][ks] ----
  // B[k][n] with n = G*64 + w*16 + ln.  ks 0,1: W_hh^T ; ks 2: W_ih^T (k<16), else 0.
  // 16-bit 32x16 B layout: lane ln = column n; elems 0..7 = K = ks*32+lg*8+i,
  // elems 8..15 = K = ks*32+16+lg*8+i.
  v16bf Bf[4][3];
#pragma unroll
  for (int G = 0; G < 4; ++G) {
    const int n = G * NH + w * 16 + ln;
#pragma unroll
    for (int ks = 0; ks < 2; ++ks) {
      const int k0 = ks * 32 + lg * 8;
      v16bf f;
#pragma unroll
      for (int i = 0; i < 8; ++i) {
        f[i]     = (__bf16)W_hh[n * NH + k0 + i];
        f[8 + i] = (__bf16)W_hh[n * NH + k0 + 16 + i];
      }
      Bf[G][ks] = f;
    }
    {   // ks == 2: x contribution, K-local = lg*8+i in [0,16), upper half zero
      v16bf f;
#pragma unroll
      for (int i = 0; i < 8; ++i) {
        f[i]     = (__bf16)W_ih[n * 16 + lg * 8 + i];
        f[8 + i] = (__bf16)0.0f;
      }
      Bf[G][2] = f;
    }
  }

  // ---- recurrent state (lane-local, matches f32 C/D layout) ----
  float c[8], h[8];
#pragma unroll
  for (int r = 0; r < 8; ++r) { c[r] = 0.0f; h[r] = 0.0f; }

  const v8bf zero8 = (v8bf)((__bf16)0.0f);

  for (int t = 0; t < T; ++t) {
    __syncthreads();   // S1: prev h/out-reset done, prev frag reads done

    // stage x[:, t, :] into LDS cols 64..79 (bf16), 2 elems per thread
    {
#pragma unroll
      for (int u = 0; u < 2; ++u) {
        int e  = tid * 2 + u;            // 0..255
        int rr = e >> 4, cc = e & 15;
        hx[rr * HXS + 64 + cc] =
            (__bf16)x[((size_t)(b0 + rr) * T + t) * 16 + cc];
      }
      if (t + 8 < T && tid < 16)          // global_prefetch_b8 for x[t+8]
        __builtin_prefetch(&x[((size_t)(b0 + tid) * T + (t + 8)) * 16], 0, 1);
    }

    __syncthreads();   // S2: x + h visible

    // A fragments from LDS: A[m][k], m = ln; elems as in 16-bit 16x32 layout.
    v16bf Af[3];
#pragma unroll
    for (int ks = 0; ks < 2; ++ks) {
      const int k0 = ks * 32 + lg * 8;
      v8bf lo = *reinterpret_cast<const v8bf*>(&hx[ln * HXS + k0]);
      v8bf hi = *reinterpret_cast<const v8bf*>(&hx[ln * HXS + k0 + 16]);
      Af[ks] = pack16(lo, hi);
    }
    {   // ks == 2: x cols 64..79, upper K half zero
      v8bf lo = *reinterpret_cast<const v8bf*>(&hx[ln * HXS + 64 + lg * 8]);
      Af[2] = pack16(lo, zero8);
    }

    // gates = bias + [h|x] @ B : 4 tiles x 3 k-steps of WMMA bf16
    v8f acc[4];
#pragma unroll
    for (int G = 0; G < 4; ++G) {
      v8f a;
#pragma unroll
      for (int r = 0; r < 8; ++r) a[r] = bias4[G];
#pragma unroll
      for (int ks = 0; ks < 3; ++ks)
        a = __builtin_amdgcn_wmma_f32_16x16x32_bf16(
                false, Af[ks], false, Bf[G][ks], (short)0, a, false, false);
      acc[G] = a;
    }

    // activations + state update (PyTorch gate order i, f, g, o)
#pragma unroll
    for (int r = 0; r < 8; ++r) {
      float iv = fast_sigmoid(acc[0][r]);
      float fv = fast_sigmoid(acc[1][r]);
      float gv = fast_tanh(acc[2][r]);
      float ov = fast_sigmoid(acc[3][r]);
      c[r] = fv * c[r] + iv * gv;
      h[r] = ov * fast_tanh(c[r]);
    }

    __syncthreads();   // S3: all A-fragment reads done before overwriting h

    // write new h (bf16) to LDS cols 0..63: row m = lg*8+r, col j = w*16+ln
#pragma unroll
    for (int r = 0; r < 8; ++r)
      hx[(lg * 8 + r) * HXS + w * 16 + ln] = (__bf16)h[r];

    __syncthreads();   // S4: h visible for output reduction

    // output head: out[b, t, o] = sum_j h[b][j] * W_out[o][j] + b_out[o]
    // lane (m = tid&15, o = po, q = pq) sums 16 j's, atomically folds into LDS.
    {
      const int m = tid & 15;
      const __bf16* hrow = &hx[m * HXS + pq * 16];
      float p = 0.0f;
#pragma unroll
      for (int i = 0; i < 16; ++i) p += (float)hrow[i] * wr[i];
      atomicAdd(&outAcc[m * 2 + po], p);   // ds_add_f32, 4-way contention
    }

    __syncthreads();   // S5: partials complete

    if (tid < 32) {
      const int m = tid & 15, o = tid >> 4;
      out[((size_t)(b0 + m) * T + t) * 2 + o] = outAcc[m * 2 + o] + bo;
      outAcc[m * 2 + o] = 0.0f;
    }
  }
}

extern "C" void kernel_launch(void* const* d_in, const int* in_sizes, int n_in,
                              void* d_out, int out_size, void* d_ws, size_t ws_size,
                              hipStream_t stream) {
  const float* x    = (const float*)d_in[0];
  const float* W_ih = (const float*)d_in[1];
  const float* W_hh = (const float*)d_in[2];
  const float* b_ih = (const float*)d_in[3];
  const float* b_hh = (const float*)d_in[4];
  const float* W_out= (const float*)d_in[5];
  const float* b_out= (const float*)d_in[6];
  float* out = (float*)d_out;

  const int B = 256, I = 16;
  int T = in_sizes[0] / (B * I);   // 2048 for the reference shapes

  dim3 grid(B / BT);               // 16 persistent workgroups
  dim3 block(128);                 // 4 wave32s
  lstm_head_wmma<<<grid, block, 0, stream>>>(x, W_ih, W_hh, b_ih, b_hh,
                                             W_out, b_out, out, T);
}
